// Attention_34772055228779
// MI455X (gfx1250) — matrix operand order repaired and checked by reading, not compiled, not run on hardware
//
#include <hip/hip_runtime.h>

// MI455X / gfx1250 implementation.
// Dominant op: scores[n,k] = sum_{d,e} x[n,d] * W_bi[k,d,e] * x[n,e]
//   = sum_d x[n,d] * ( X_f16 @ W_bi[k,d,:]^T )  -> fp16 WMMA 16x16x32, fp32 acc.
// W_bi (512 MB fp32) is the streamed operand, DMA'd global->LDS with the
// CDNA5 Tensor Data Mover (TENSORcnt), read ~once (L2 catches the 4x n-block
// reuse) => ~23 us HBM traffic at 23.3 TB/s, matched by ~27 us of fp16 WMMA.
// d-reduction split 4 ways (separate partial buffers, no atomics -> fully
// deterministic) for WGP occupancy; softmax sums the partials on read.

#define N_CTX   512
#define D_MODEL 512

typedef __attribute__((ext_vector_type(16))) _Float16 v16h;
typedef __attribute__((ext_vector_type(8)))  _Float16 v8h;
typedef __attribute__((ext_vector_type(8)))  float    v8f;
typedef __attribute__((ext_vector_type(4)))  float    v4f;
typedef __attribute__((ext_vector_type(4)))  unsigned int v4u;
typedef __attribute__((ext_vector_type(8)))  int      v8i;
typedef __attribute__((ext_vector_type(4)))  int      v4i;

union AB16 { v16h v; _Float16 h[16]; };
union F8   { v8f  v; float    f[8];  };

#define XSTR  520   // fp16 elems per x row in LDS (512 + 8 pad)
#define WSTR  520   // fp32 elems per W row in LDS: 512 + 4dw pad after each 256dw
                    // (matches TDM pad_interval=256dw / pad_amount=4dw; keeps
                    //  every 16-float fragment run 16B aligned)
#define NB    128   // n rows per workgroup (8 waves x 16-row tile each)
#define KT    16    // k columns per workgroup
#define SPLIT 4     // split-K over d
#define DSEG  (D_MODEL / SPLIT)

#if __has_builtin(__builtin_amdgcn_tensor_load_to_lds) && __has_builtin(__builtin_amdgcn_s_wait_tensorcnt)
#define USE_TDM 1
#else
#define USE_TDM 0
#endif

// padded LDS dword offset for W element (k, e): cumulative pos k*512+e, +4dw per 256dw
__device__ __forceinline__ int widx(int k, int e) {
  return k * WSTR + e + ((e >> 8) << 2);
}

#if USE_TDM
// Issue a TDM DMA of W[k0..k0+15][d][0..511] (fp32) into LDS at ldst, writing
// the padded layout directly. D# per CDNA5 ISA ch.8 (2D tile, groups 2/3 zero).
// This toolchain exposes the 6-arg builtin form:
//   (uint32x4 g0, int32x8 g1, int32x4 g2, int32x4 g3, int32x8 g4, i32 cpol)
__device__ __forceinline__ void tdm_stage_w(const float* gsrc, void* ldst) {
  unsigned long long ga = (unsigned long long)(uintptr_t)gsrc;
  unsigned int       la = (unsigned int)(uintptr_t)ldst;   // addr[31:0] = LDS offset
  v4u g0;
  g0[0] = 1u;                                   // count=1 (valid), user desc
  g0[1] = la;                                   // lds_addr
  g0[2] = (unsigned int)ga;                     // global_addr[31:0]
  g0[3] = (unsigned int)((ga >> 32) & 0x01FFFFFFu) | (2u << 30); // addr[56:32], type=2
  v8i g1;
  g1[0] = (int)((2u << 16)     // data_size = 4B
              | (1u << 20)     // pad_enable
              | (7u << 22)     // pad_interval code 7 = 256 dwords
              | (3u << 25));   // pad_amount  code 3 = 4 dwords
  g1[1] = (int)(512u << 16);   // tensor_dim0 = 512 (bits[79:48] low half)
  g1[2] = (int)(16u  << 16);   // tensor_dim1 = 16  (bits[111:80] low half)
  g1[3] = (int)(512u << 16);   // tile_dim0 = 512   (bits[127:112])
  g1[4] = (int)16u;            // tile_dim1 = 16, tile_dim2 = 0
  g1[5] = (int)(512u * 512u);  // tensor_dim0_stride = 262144 elems (k-row stride)
  g1[6] = 0;
  g1[7] = 0;
  v4i g2 = {0, 0, 0, 0};       // 2D tensor: groups 2/3 unused
  v4i g3 = {0, 0, 0, 0};
  v8i g4 = {0, 0, 0, 0, 0, 0, 0, 0};
  __builtin_amdgcn_tensor_load_to_lds(g0, g1, g2, g3, g4, 0);
}
#endif

__global__ __launch_bounds__(256)
void bilinear_scores_wmma(const float* __restrict__ x,
                          const float* __restrict__ Wbi,
                          float* __restrict__ parts) {
  __shared__ _Float16 xlds[NB * XSTR];          // ~130 KB
  __shared__ float    wlds[2][KT * WSTR];       // ~67 KB double buffer

  const int t     = threadIdx.x;
  const int lane  = t & 31;
  const int wv    = t >> 5;                     // wave id 0..7 -> n-tile
  const int nBase = blockIdx.x * NB;            // n-block (fast grid dim: L2 reuse of W)
  const int k0    = blockIdx.y * KT;            // k-tile
  const int d0    = blockIdx.z * DSEG;          // split-K segment

  // ---- stage x block: fp32 global -> fp16 LDS (reused for all d in segment) ----
  for (int j = 0; j < 64; ++j) {
    int idx = t + j * 256;                      // float4 index within 128x512 block
    int row = idx >> 7;
    int f4  = idx & 127;
    v4f xv = *(const v4f*)(x + (size_t)(nBase + row) * D_MODEL + f4 * 4);
    _Float16* dst = &xlds[row * XSTR + f4 * 4];
    dst[0] = (_Float16)xv.x; dst[1] = (_Float16)xv.y;
    dst[2] = (_Float16)xv.z; dst[3] = (_Float16)xv.w;
  }

  // ---- stage W slice d0 into buffer 0 ----
#if USE_TDM
  if (wv == 0) {
    tdm_stage_w(Wbi + ((size_t)k0 * D_MODEL + d0) * D_MODEL, &wlds[0][0]);
    __builtin_amdgcn_s_wait_tensorcnt(0);
  }
#else
  for (int j = 0; j < 8; ++j) {
    int idx = t + j * 256;                      // float4 index within 16x512 slice
    int kk  = idx >> 7;
    int f4  = idx & 127;
    v4f wv4 = *(const v4f*)(Wbi + ((size_t)(k0 + kk) * D_MODEL + d0) * D_MODEL + f4 * 4);
    *(v4f*)&wlds[0][widx(kk, f4 * 4)] = wv4;
  }
#endif
  __syncthreads();

  const int colSel = lane >> 4;                 // 0: lanes 0-15, 1: lanes 16-31
  const int rowM   = lane & 15;                 // A row / B col / D col index
  const int aOff   = colSel * 16;               // A K-offset (contiguous, matches B)
  const int bOff   = colSel * 16;               // B K-offset (contiguous, matches A)

  F8 acc;
  #pragma unroll
  for (int v = 0; v < 8; ++v) acc.f[v] = 0.0f;

  const _Float16* xrow = &xlds[(wv * 16 + rowM) * XSTR];

  for (int dd = 0; dd < DSEG; ++dd) {
    const int  d    = d0 + dd;
    const int  cur  = dd & 1;
    const bool more = (dd + 1) < DSEG;

#if USE_TDM
    // Kick the DMA for the next slice; it runs while 8 waves do 128 WMMAs.
    if (wv == 0 && more)
      tdm_stage_w(Wbi + ((size_t)k0 * D_MODEL + (d + 1)) * D_MODEL, &wlds[cur ^ 1][0]);
#else
    v4f st[8];
    if (more) {
      #pragma unroll
      for (int j = 0; j < 8; ++j) {
        int idx = t + j * 256;
        int kk  = idx >> 7;
        int f4  = idx & 127;
        st[j] = *(const v4f*)(Wbi + ((size_t)(k0 + kk) * D_MODEL + (d + 1)) * D_MODEL + f4 * 4);
      }
    }
#endif

    const _Float16 xd = xrow[d];                // per-lane row scalar x[n, d]
    const float* wrow = &wlds[cur][rowM * WSTR];

    for (int ec = 0; ec < 16; ++ec) {
      const int e0 = ec * 32;
      const int eb = e0 + bOff;

      // B fragment: col k = rowM, K = eb + j  (fp32 LDS -> fp16), pad-adjusted
      AB16 bf;
      const float* wp = wrow + eb + ((eb >> 8) << 2);
      #pragma unroll
      for (int j = 0; j < 16; ++j) bf.h[j] = (_Float16)wp[j];

      // A fragment: row m = rowM, K = e0+aOff..+15 (contiguous),
      // scaled by the per-row scalar x[n,d] (diag trick).
      v8h p0 = *(const v8h*)(xrow + e0 + aOff);
      v8h p1 = *(const v8h*)(xrow + e0 + aOff + 8);
      AB16 af;
      #pragma unroll
      for (int j = 0; j < 8; ++j) { af.h[j] = xd * p0[j]; af.h[8 + j] = xd * p1[j]; }

      acc.v = __builtin_amdgcn_wmma_f32_16x16x32_f16(
          false, af.v, false, bf.v, (short)0, acc.v, false, false);
    }

#if USE_TDM
    if (wv == 0) __builtin_amdgcn_s_wait_tensorcnt(0);
#else
    if (more) {
      const int nxt = cur ^ 1;
      #pragma unroll
      for (int j = 0; j < 8; ++j) {
        int idx = t + j * 256;
        int kk  = idx >> 7;
        int f4  = idx & 127;
        *(v4f*)&wlds[nxt][widx(kk, f4 * 4)] = st[j];
      }
    }
#endif
    __syncthreads();
  }

  // D layout: VGPR v, lanes<16 -> M=v, lanes>=16 -> M=v+8; N = lane&15.
  // Exclusive tile per (block,z) -> plain stores, fully deterministic.
  float* sp = parts + (size_t)blockIdx.z * N_CTX * N_CTX;
  #pragma unroll
  for (int v = 0; v < 8; ++v) {
    int row = nBase + wv * 16 + v + colSel * 8;
    sp[(size_t)row * N_CTX + k0 + rowM] = acc.f[v];
  }
}

// Causal softmax over dim=1. Sums the SPLIT partial score buffers on read,
// writes normalized attn into partial 0. One 256-thread block per row.
__global__ __launch_bounds__(256)
void softmax_causal_inplace(float* __restrict__ parts) {
  const int row = blockIdx.x;
  const int t   = threadIdx.x;
  __shared__ float red[256];
  const size_t P = (size_t)N_CTX * N_CTX;
  float* r = parts + (size_t)row * N_CTX;

  float s0 = 0.f, s1 = 0.f;
  #pragma unroll
  for (int z = 0; z < SPLIT; ++z) {
    s0 += r[z * P + t];
    s1 += r[z * P + t + 256];
  }

  const float NEG = -__builtin_inff();
  float v0 = (t       <= row) ? s0 : NEG;
  float v1 = (t + 256 <= row) ? s1 : NEG;

  red[t] = fmaxf(v0, v1); __syncthreads();
  for (int s = 128; s > 0; s >>= 1) { if (t < s) red[t] = fmaxf(red[t], red[t + s]); __syncthreads(); }
  float rmax = red[0]; __syncthreads();

  float e0 = (t       <= row) ? __expf(v0 - rmax) : 0.f;
  float e1 = (t + 256 <= row) ? __expf(v1 - rmax) : 0.f;
  red[t] = e0 + e1; __syncthreads();
  for (int s = 128; s > 0; s >>= 1) { if (t < s) red[t] += red[t + s]; __syncthreads(); }
  float inv = 1.0f / red[0];

  r[t]       = e0 * inv;
  r[t + 256] = e1 * inv;
}

// Small fp32 tiled 512x512x512 GEMM. TB=1 -> C = A * B^T.
template <int TB>
__global__ __launch_bounds__(256)
void gemm512(const float* __restrict__ A, const float* __restrict__ B,
             float* __restrict__ C) {
  __shared__ float As[16][17];
  __shared__ float Bs[16][17];
  const int tx  = threadIdx.x, ty = threadIdx.y;
  const int col = blockIdx.x * 16 + tx;
  const int row = blockIdx.y * 16 + ty;
  float acc = 0.f;
  for (int kt = 0; kt < 512; kt += 16) {
    As[ty][tx] = A[(size_t)row * 512 + kt + tx];
    if (TB) Bs[ty][tx] = B[(size_t)(blockIdx.x * 16 + tx) * 512 + kt + ty];
    else    Bs[ty][tx] = B[(size_t)(kt + ty) * 512 + col];
    __syncthreads();
    #pragma unroll
    for (int k = 0; k < 16; ++k) acc += As[ty][k] * Bs[k][tx];
    __syncthreads();
  }
  C[(size_t)row * 512 + col] = acc;
}

extern "C" void kernel_launch(void* const* d_in, const int* in_sizes, int n_in,
                              void* d_out, int out_size, void* d_ws, size_t ws_size,
                              hipStream_t stream) {
  const float* x    = (const float*)d_in[0];   // [512, 512]
  const float* Wbi  = (const float*)d_in[1];   // [512, 512, 512]
  const float* Wout = (const float*)d_in[2];   // [512, 512]
  float* out   = (float*)d_out;                // [512, 512]
  float* parts = (float*)d_ws;                 // SPLIT x 1 MB partial scores
  float* attn  = parts;                        // softmax writes attn into partial 0
  float* tmp   = parts + (size_t)SPLIT * N_CTX * N_CTX;  // 1 MB

  bilinear_scores_wmma<<<dim3(N_CTX / NB, N_CTX / KT, SPLIT), 256, 0, stream>>>(x, Wbi, parts);
  softmax_causal_inplace<<<N_CTX, 256, 0, stream>>>(parts);
  gemm512<0><<<dim3(32, 32), dim3(16, 16), 0, stream>>>(attn, x, tmp);   // attn @ x
  gemm512<1><<<dim3(32, 32), dim3(16, 16), 0, stream>>>(tmp, Wout, out); // @ W_out^T
}
